// CrossMultiHeadAttention_74861279969599
// MI455X (gfx1250) — matrix operand order repaired and checked by reading, not compiled, not run on hardware
//
#include <hip/hip_runtime.h>
#include <cstdint>
#include <cstddef>

// ---------------------------------------------------------------------------
// CrossMultiHeadAttention for MI455X (gfx1250, wave32, WMMA + async-LDS DMA).
// B=8, S=1024, D=1024, H=16, hd=64. bf16 WMMA operands, f32 accumulation.
// ---------------------------------------------------------------------------

typedef __bf16 bf16;
typedef __attribute__((ext_vector_type(16))) __bf16 v16bf;
typedef __attribute__((ext_vector_type(8)))  __bf16 v8bf;
typedef __attribute__((ext_vector_type(4)))  __bf16 v4bf;
typedef __attribute__((ext_vector_type(8)))  float  v8f;

union Frag {
    v16bf v;
    v8bf  h[2];
};

__device__ __forceinline__ v8f wmma_bf16(const Frag& a, const Frag& b, v8f c) {
    // 8 args: (neg_a, A, neg_b, B, c_mod, C, reuse_a, reuse_b)
    return __builtin_amdgcn_wmma_f32_16x16x32_bf16(
        false, a.v, false, b.v, (short)0, c, false, false);
}

// gfx1250 async HBM->LDS DMA (16B per lane), tracked by ASYNCcnt.
__device__ __forceinline__ void async_copy_b128(uint32_t lds_byte_addr,
                                                const void* gptr) {
    uint64_t ga = (uint64_t)(uintptr_t)gptr;
    asm volatile("global_load_async_to_lds_b128 %0, %1, off"
                 :: "v"(lds_byte_addr), "v"(ga) : "memory");
}
__device__ __forceinline__ void wait_asynccnt_le3() {
    asm volatile("s_wait_asynccnt 0x3" ::: "memory");
}
__device__ __forceinline__ void wait_asynccnt_0() {
    asm volatile("s_wait_asynccnt 0x0" ::: "memory");
}

// Problem constants
#define BATCH 8
#define SEQ   1024
#define DIM   1024
#define HEADS 16
#define HDIM  64

// ---------------------------------------------------------------------------
// fp32 -> bf16 bulk conversion (vectorized float4 -> 4x bf16)
// ---------------------------------------------------------------------------
__global__ __launch_bounds__(256) void cvt_f32_bf16_kernel(
    const float* __restrict__ in, bf16* __restrict__ out, int n4)
{
    int i = blockIdx.x * 256 + threadIdx.x;
    if (i < n4) {
        float4 v = ((const float4*)in)[i];
        v4bf t = {(bf16)v.x, (bf16)v.y, (bf16)v.z, (bf16)v.w};
        ((v4bf*)out)[i] = t;
    }
}

// ---------------------------------------------------------------------------
// Tiled bf16 GEMM with async-LDS double buffering.
// C[M,N] = A[M,K] * B[K,N] + bias; A,B bf16 in HBM.
// Block tile 128x64, BK=32, 8 waves; wave strip 16(M) x 64(N) = 4 accums.
// Tiles staged by GLOBAL_LOAD_ASYNC_TO_LDS_B128 (3 per thread per tile:
// A 128x32 = 512 x16B chunks -> 2/thread; B 32x64 = 256 chunks -> 1/thread).
// MODE 0: QK-fused epilogue -> out_q/out_k bf16 [B,H,S,hd]
// MODE 1: V epilogue        -> out_q (=V) bf16 [B,H,S,hd]
// MODE 2: plain fp32        -> out_f[M,N]
// ---------------------------------------------------------------------------
template <int MODE>
__global__ __launch_bounds__(256) void gemm_bf16_kernel(
    const bf16* __restrict__ A, const bf16* __restrict__ Bm,
    const float* __restrict__ bias,
    bf16* __restrict__ out_q, bf16* __restrict__ out_k,
    float* __restrict__ out_f,
    int M, int N, int Kd)
{
    constexpr int BM = 128, BN = 64, BK = 32;
    constexpr int LDA = 40;            // elems/row (80B, 16B-multiple)
    constexpr int LDB = 72;            // elems/row (144B, 16B-multiple)
    constexpr int ASZ = BM * LDA;      // 5120 elems = 10240B
    constexpr int BSZ = BK * LDB;      // 2304 elems = 4608B

    __shared__ __align__(16) bf16 sm[2 * ASZ + 2 * BSZ];

    const int tid  = threadIdx.x;
    const int w    = tid >> 5;
    const int lane = tid & 31;
    const int r0   = blockIdx.y * BM;
    const int n0   = blockIdx.x * BN;

    const uint32_t smBase = (uint32_t)(uintptr_t)(&sm[0]);  // LDS byte offset
    const uint32_t aBase0 = smBase;
    const uint32_t bBase0 = smBase + (uint32_t)(2 * ASZ * 2);

    // A-fragment addressing: lane L -> row L%16; k runs [kb..kb+7],[kb+16..kb+23]
    const int mloc  = w * 16 + (lane & 15);
    const int kbase = (lane & 16) ? 8 : 0;

    v8f acc[4];
#pragma unroll
    for (int g = 0; g < 4; ++g)
#pragma unroll
        for (int r = 0; r < 8; ++r) acc[g][r] = 0.f;

    // async issue of one k-tile into buffer `buf`
    auto issue_tile = [&](int k0, int buf) {
        const uint32_t aB = aBase0 + (uint32_t)(buf * ASZ * 2);
        const uint32_t bB = bBase0 + (uint32_t)(buf * BSZ * 2);
#pragma unroll
        for (int i = 0; i < 2; ++i) {            // A: 512 chunks / 256 thr
            int idx = tid + 256 * i;
            int row = idx >> 2, seg = idx & 3;
            async_copy_b128(aB + (uint32_t)((row * LDA + seg * 8) * 2),
                            A + (size_t)(r0 + row) * Kd + k0 + seg * 8);
        }
        {                                        // B: 256 chunks / 256 thr
            int row = tid >> 3, seg = tid & 7;
            async_copy_b128(bB + (uint32_t)((row * LDB + seg * 8) * 2),
                            Bm + (size_t)(k0 + row) * N + n0 + seg * 8);
        }
    };

    issue_tile(0, 0);
    const int NT = Kd / BK;
    for (int t = 0; t < NT; ++t) {
        const int buf = t & 1;
        if (t + 1 < NT) {
            __syncthreads();                 // everyone done reading buf^1
            issue_tile((t + 1) * BK, buf ^ 1);
            wait_asynccnt_le3();             // tile t landed (in-order), t+1 in flight
        } else {
            wait_asynccnt_0();
        }
        __syncthreads();                     // tile t visible block-wide

        const bf16* Abuf = sm + buf * ASZ;
        const bf16* Bbuf = sm + 2 * ASZ + buf * BSZ;

        Frag a;
        a.h[0] = *(const v8bf*)(&Abuf[mloc * LDA + kbase]);
        a.h[1] = *(const v8bf*)(&Abuf[mloc * LDA + kbase + 16]);
#pragma unroll
        for (int g = 0; g < 4; ++g) {
            Frag b;  // B 32x16: lane -> column lane%16; k in table order
#pragma unroll
            for (int s = 0; s < 8; ++s) {
                b.v[s]     = Bbuf[(kbase + s) * LDB + g * 16 + (lane & 15)];
                b.v[s + 8] = Bbuf[(kbase + 16 + s) * LDB + g * 16 + (lane & 15)];
            }
            acc[g] = wmma_bf16(a, b, acc[g]);
        }
    }

    // C layout: lanes 0-15 -> (M=r, N=lane); lanes 16-31 -> (M=8+r, N=lane-16)
    const int prow = (lane & 16) ? 8 : 0;
    const int ncol = lane & 15;

#pragma unroll
    for (int g = 0; g < 4; ++g) {
        const int   c    = n0 + g * 16 + ncol;
        const float bval = bias[c];
        if (MODE == 0) {
            const int  head = c >> 7;
            const int  off  = c & 127;
            bf16*      dst  = (off < 64) ? out_q : out_k;
            const int  o    = off & 63;
#pragma unroll
            for (int r = 0; r < 8; ++r) {
                int m = r0 + w * 16 + prow + r;
                int b = m >> 10, s = m & (SEQ - 1);
                dst[(((size_t)(b * HEADS + head)) * SEQ + s) * HDIM + o] =
                    (bf16)(acc[g][r] + bval);
            }
        } else if (MODE == 1) {
            const int head = c >> 6;
            const int o    = c & 63;
#pragma unroll
            for (int r = 0; r < 8; ++r) {
                int m = r0 + w * 16 + prow + r;
                int b = m >> 10, s = m & (SEQ - 1);
                out_q[(((size_t)(b * HEADS + head)) * SEQ + s) * HDIM + o] =
                    (bf16)(acc[g][r] + bval);
            }
        } else {
#pragma unroll
            for (int r = 0; r < 8; ++r) {
                int m = r0 + w * 16 + prow + r;
                out_f[(size_t)m * N + c] = acc[g][r] + bval;
            }
        }
    }
}

// ---------------------------------------------------------------------------
// Flash attention, bf16 WMMA, online softmax. grid = (S/64, B*H), 4 waves,
// wave = 16 query rows; kv blocks of 32 keys:
//   S(16x32) = Q @ K^T : 4 WMMAs     O += P(16x32) @ V(32x64) : 4 WMMAs
// ---------------------------------------------------------------------------
__global__ __launch_bounds__(128) void flash_attn_kernel(
    const bf16* __restrict__ Qg, const bf16* __restrict__ Kg,
    const bf16* __restrict__ Vg, bf16* __restrict__ Og)
{
    constexpr int   KT_LD = 40;
    constexpr int   P_LD  = 40;
    constexpr float SCALE = 0.125f;  // 1/sqrt(64)

    __shared__ __align__(16) bf16 Kt[64 * KT_LD];     // K^T tile: [dim][key]
    __shared__ __align__(16) bf16 Pld[4][16 * P_LD];  // per-wave P scratch

    const int w    = threadIdx.x >> 5;
    const int lane = threadIdx.x & 31;
    const int bh   = blockIdx.y;
    const int q0   = blockIdx.x * 64 + w * 16;

    const bf16* Qb = Qg + (size_t)bh * SEQ * HDIM;
    const bf16* Kb = Kg + (size_t)bh * SEQ * HDIM;
    const bf16* Vb = Vg + (size_t)bh * SEQ * HDIM;

    const int kbase = (lane & 16) ? 8 : 0;
    const int prow  = (lane & 16) ? 8 : 0;
    const int ncol  = lane & 15;

    Frag aq[2];
#pragma unroll
    for (int ks = 0; ks < 2; ++ks) {
        const bf16* p = Qb + (size_t)(q0 + (lane & 15)) * HDIM + ks * 32 + kbase;
        aq[ks].h[0] = *(const v8bf*)p;
        aq[ks].h[1] = *(const v8bf*)(p + 16);
    }

    float mrow[8], lrow[8];
    v8f   oacc[4];
#pragma unroll
    for (int r = 0; r < 8; ++r) { mrow[r] = -3.0e38f; lrow[r] = 0.f; }
#pragma unroll
    for (int g = 0; g < 4; ++g)
#pragma unroll
        for (int r = 0; r < 8; ++r) oacc[g][r] = 0.f;

    for (int kb = 0; kb < SEQ / 32; ++kb) {
        const int kv0 = kb * 32;
        __syncthreads();
        {   // transpose K block [32 keys][64 dims] -> Kt[dim][key]
            const int   key = threadIdx.x & 31;
            const int   seg = threadIdx.x >> 5;
            const bf16* kp  = Kb + (size_t)(kv0 + key) * HDIM + seg * 16;
            v8bf k0 = *(const v8bf*)kp;
            v8bf k1 = *(const v8bf*)(kp + 8);
#pragma unroll
            for (int j = 0; j < 8; ++j) {
                Kt[(seg * 16 + j) * KT_LD + key]       = k0[j];
                Kt[(seg * 16 + 8 + j) * KT_LD + key]   = k1[j];
            }
            if (kb + 1 < SEQ / 32) {   // prefetch next K/V blocks into caches
                __builtin_prefetch(Kb + (size_t)(kv0 + 32 + key) * HDIM, 0, 0);
                __builtin_prefetch(Vb + (size_t)(kv0 + 32 + key) * HDIM, 0, 0);
            }
        }
        __syncthreads();

        // ----- scores -----
        v8f sc[2];
#pragma unroll
        for (int g2 = 0; g2 < 2; ++g2)
#pragma unroll
            for (int r = 0; r < 8; ++r) sc[g2][r] = 0.f;
#pragma unroll
        for (int g2 = 0; g2 < 2; ++g2) {
#pragma unroll
            for (int ks = 0; ks < 2; ++ks) {
                Frag bk;
                const bf16* p = Kb + (size_t)(kv0 + g2 * 16 + ncol) * HDIM + ks * 32 + kbase;
                bk.h[0] = *(const v8bf*)p;
                bk.h[1] = *(const v8bf*)(p + 16);
                sc[g2] = wmma_bf16(aq[ks], bk, sc[g2]);
            }
        }

        // ----- online softmax -----
        float alpha[8];
#pragma unroll
        for (int r = 0; r < 8; ++r) {
            float s0 = fmaxf(sc[0][r], sc[1][r]) * SCALE;
#pragma unroll
            for (int off = 1; off < 16; off <<= 1)
                s0 = fmaxf(s0, __shfl_xor(s0, off, 32));
            float mn = fmaxf(mrow[r], s0);
            alpha[r] = __expf(mrow[r] - mn);
            mrow[r]  = mn;
        }
#pragma unroll
        for (int r = 0; r < 8; ++r) {
            float sum = 0.f;
#pragma unroll
            for (int g2 = 0; g2 < 2; ++g2) {
                float pv = __expf(sc[g2][r] * SCALE - mrow[r]);
                Pld[w][(prow + r) * P_LD + g2 * 16 + ncol] = (bf16)pv;
                sum += pv;
            }
#pragma unroll
            for (int off = 1; off < 16; off <<= 1)
                sum += __shfl_xor(sum, off, 32);
            lrow[r] = lrow[r] * alpha[r] + sum;
        }
#pragma unroll
        for (int g = 0; g < 4; ++g)
#pragma unroll
            for (int r = 0; r < 8; ++r) oacc[g][r] *= alpha[r];

        asm volatile("s_wait_dscnt 0x0" ::: "memory");  // wave-local P fence

        Frag pa;
        {
            const bf16* pp = &Pld[w][(lane & 15) * P_LD + kbase];
            pa.h[0] = *(const v8bf*)pp;
            pa.h[1] = *(const v8bf*)(pp + 16);
        }
#pragma unroll
        for (int g = 0; g < 4; ++g) {
            Frag bv;
            const bf16* vp = Vb + (size_t)kv0 * HDIM + g * 16 + ncol;
#pragma unroll
            for (int s = 0; s < 8; ++s) {
                bv.v[s]     = vp[(size_t)(kbase + s) * HDIM];
                bv.v[s + 8] = vp[(size_t)(kbase + 16 + s) * HDIM];
            }
            oacc[g] = wmma_bf16(pa, bv, oacc[g]);
        }
    }

    // ----- epilogue: O /= l, bf16 [B,S,D] with heads re-merged -----
    const int b = bh >> 4, h = bh & 15;
#pragma unroll
    for (int g = 0; g < 4; ++g) {
#pragma unroll
        for (int r = 0; r < 8; ++r) {
            int srow = q0 + prow + r;
            int col  = h * HDIM + g * 16 + ncol;
            Og[((size_t)(b * SEQ + srow)) * DIM + col] =
                (bf16)(oacc[g][r] / lrow[r]);
        }
    }
}

// ---------------------------------------------------------------------------
// Launch. d_in: input_query, encoding_output, Wqk, bqk, Wv, bv, Wo, bo
// ws: encb 16MB | inpb 16MB | wqkb 4MB | wvb 2MB | wob 2MB |
//     Q 16MB | K 16MB | V 16MB | Att(bf16) 16MB  => 104MB
// ---------------------------------------------------------------------------
extern "C" void kernel_launch(void* const* d_in, const int* in_sizes, int n_in,
                              void* d_out, int out_size, void* d_ws, size_t ws_size,
                              hipStream_t stream) {
    const float* inp = (const float*)d_in[0];
    const float* enc = (const float*)d_in[1];
    const float* Wqk = (const float*)d_in[2];
    const float* bqk = (const float*)d_in[3];
    const float* Wv  = (const float*)d_in[4];
    const float* bv  = (const float*)d_in[5];
    const float* Wo  = (const float*)d_in[6];
    const float* bo  = (const float*)d_in[7];
    float*       out = (float*)d_out;

    const size_t MB = 1024 * 1024;
    char* ws   = (char*)d_ws;
    bf16* encb = (bf16*)(ws);
    bf16* inpb = (bf16*)(ws + 16 * MB);
    bf16* wqkb = (bf16*)(ws + 32 * MB);
    bf16* wvb  = (bf16*)(ws + 36 * MB);
    bf16* wob  = (bf16*)(ws + 38 * MB);
    bf16* Qws  = (bf16*)(ws + 40 * MB);
    bf16* Kws  = (bf16*)(ws + 56 * MB);
    bf16* Vws  = (bf16*)(ws + 72 * MB);
    bf16* Att  = (bf16*)(ws + 88 * MB);

    const int M = BATCH * SEQ;  // 8192

    // 0) one-shot fp32 -> bf16 conversions (compute-bound problem; ~3us of BW)
    cvt_f32_bf16_kernel<<<dim3((M * DIM / 4) / 256), 256, 0, stream>>>(enc, encb, M * DIM / 4);
    cvt_f32_bf16_kernel<<<dim3((M * DIM / 4) / 256), 256, 0, stream>>>(inp, inpb, M * DIM / 4);
    cvt_f32_bf16_kernel<<<dim3((DIM * 2 * DIM / 4) / 256), 256, 0, stream>>>(Wqk, wqkb, DIM * 2 * DIM / 4);
    cvt_f32_bf16_kernel<<<dim3((DIM * DIM / 4) / 256), 256, 0, stream>>>(Wv, wvb, DIM * DIM / 4);
    cvt_f32_bf16_kernel<<<dim3((DIM * DIM / 4) / 256), 256, 0, stream>>>(Wo, wob, DIM * DIM / 4);

    // 1) fused QK projection -> Q/K bf16 [B,H,S,hd]
    gemm_bf16_kernel<0><<<dim3((2 * DIM) / 64, M / 128), 256, 0, stream>>>(
        encb, wqkb, bqk, Qws, Kws, nullptr, M, 2 * DIM, DIM);
    // 2) V projection -> V bf16 [B,H,S,hd]
    gemm_bf16_kernel<1><<<dim3(DIM / 64, M / 128), 256, 0, stream>>>(
        inpb, wvb, bv, Vws, nullptr, nullptr, M, DIM, DIM);
    // 3) flash attention -> Att bf16 [B,S,D]
    flash_attn_kernel<<<dim3(SEQ / 64, BATCH * HEADS), 128, 0, stream>>>(
        Qws, Kws, Vws, Att);
    // 4) output projection -> d_out fp32
    gemm_bf16_kernel<2><<<dim3(DIM / 64, M / 128), 256, 0, stream>>>(
        Att, wob, bo, nullptr, nullptr, out, M, DIM, DIM);
}
